// QwenAttention_64390149701647
// MI455X (gfx1250) — matrix operand order repaired and checked by reading, not compiled, not run on hardware
//
#include <hip/hip_runtime.h>
#include <hip/hip_bf16.h>

typedef __bf16 bf16_t;
typedef __attribute__((ext_vector_type(16))) __bf16 v16bf;
typedef __attribute__((ext_vector_type(8)))  __bf16 v8bf;
typedef __attribute__((ext_vector_type(8)))  float  v8f;

#define S_LEN 2048
#define D_DIM 2048
#define H_Q   16
#define H_KV  2
#define HD_   128
#define QKV_N 2560
#define NEGF  -3.0e38f

// ---- CDNA5 Tensor Data Mover availability --------------------------------
#if defined(__gfx1250__) && __has_builtin(__builtin_amdgcn_tensor_load_to_lds) && \
    __has_builtin(__builtin_amdgcn_s_wait_tensorcnt)
#define USE_TDM 1
#if __has_include(<hip/amd_detail/amd_gfx1250_TDM.h>)
#define TDM_6ARG 1   // therock headers -> 6-arg builtin
#endif
#endif

#ifdef USE_TDM
typedef __attribute__((ext_vector_type(4))) unsigned int u32x4;
typedef __attribute__((ext_vector_type(8))) int          i32x8;
typedef __attribute__((ext_vector_type(4))) int          i32x4;

// 2D bf16 tile load: tile_dim0 = 32 elems (contiguous K), tile_dim1 = rows.
// pad_enable: +4 DWORDs after every 16 DWORDs -> LDS row stride = 40 elems.
__device__ __forceinline__ void tdm_load_tile_bf16(const bf16_t* gsrc, unsigned lds_off,
                                                   unsigned rows, unsigned row_stride) {
  unsigned long long ga = (unsigned long long)(const void*)gsrc;
  u32x4 g0 = { 1u,                                   // count=1, is_restore=0
               lds_off,                               // lds_addr
               (unsigned)ga,                          // global_addr[31:0]
               ((unsigned)(ga >> 32) & 0x01ffffffu) | (2u << 30) };  // addr[56:32] | type=2
  i32x8 g1;
  g1[0] = (1 << 16) | (1 << 20) | (3 << 22) | (3 << 25); // data_size=2B, pad_en, 16DW/pad, 4DW pad
  g1[1] = (int)(32u << 16);                              // tensor_dim0 = 32 (lo16)
  g1[2] = (int)((rows & 0xffffu) << 16);                 // tensor_dim1 (lo16)
  g1[3] = (int)(32u << 16);                              // tile_dim0 = 32
  g1[4] = (int)(rows & 0xffffu);                         // tile_dim1 = rows, tile_dim2=0
  g1[5] = (int)row_stride;                               // tensor_dim0_stride[31:0] (elems)
  g1[6] = 0;
  g1[7] = 0;
  i32x4 gz = {0, 0, 0, 0};
#ifdef TDM_6ARG
  i32x8 gz8 = {0, 0, 0, 0, 0, 0, 0, 0};
  __builtin_amdgcn_tensor_load_to_lds(g0, g1, gz, gz, gz8, 0);
#else
  __builtin_amdgcn_tensor_load_to_lds(g0, g1, gz, gz, 0);
#endif
}
#endif  // USE_TDM

// ---------------------------------------------------------------------------
__global__ __launch_bounds__(256) void k_f32_to_bf16(const float* __restrict__ in,
                                                     bf16_t* __restrict__ out, int n) {
  for (int i = blockIdx.x * 256 + threadIdx.x; i < n; i += gridDim.x * 256)
    out[i] = (bf16_t)in[i];
}

__global__ __launch_bounds__(256) void k_transpose_bf16(const float* __restrict__ in,
                                                        bf16_t* __restrict__ out,
                                                        int K, int N) {
  int total = K * N;
  for (int i = blockIdx.x * 256 + threadIdx.x; i < total; i += gridDim.x * 256) {
    int k = i / N, n = i - k * N;
    out[(size_t)n * K + k] = (bf16_t)in[i];
  }
}

// ---------------------------------------------------------------------------
// C[M][N] = A[M][K] x BT[N][K] + bias.  Block: 256 thr, tile 128x128, K-slab 32,
// TDM double-buffered LDS staging; wave tile 32x64 = 2x4 wmma tiles.
#define BM 128
#define BN 128
#define BK 32
#define LDSW 40  // padded LDS row stride in bf16 elems (64B row + 16B pad)

__global__ __launch_bounds__(256) void k_gemm_bf16(const bf16_t* __restrict__ A,
                                                   const bf16_t* __restrict__ BT,
                                                   const float* __restrict__ bias,
                                                   float* __restrict__ C,
                                                   int M, int N, int K) {
  __shared__ __align__(16) bf16_t sA[2][BM * LDSW];
  __shared__ __align__(16) bf16_t sB[2][BN * LDSW];

  const int lane = threadIdx.x & 31;
  const int wave = threadIdx.x >> 5;
  const int wm = wave & 3, wn = wave >> 2;
  const int half = lane >> 4, l16 = lane & 15;
  const int blockM = blockIdx.y * BM, blockN = blockIdx.x * BN;

  v8f acc[2][4];
#pragma unroll
  for (int a = 0; a < 2; ++a)
#pragma unroll
    for (int b = 0; b < 4; ++b) acc[a][b] = (v8f){0.f,0.f,0.f,0.f,0.f,0.f,0.f,0.f};

  auto stage = [&](int buf, int kk) {
#ifdef USE_TDM
    if (wave == 0) {
      tdm_load_tile_bf16(A  + (size_t)blockM * K + kk, (unsigned)(size_t)&sA[buf][0], BM, K);
      tdm_load_tile_bf16(BT + (size_t)blockN * K + kk, (unsigned)(size_t)&sB[buf][0], BN, K);
    }
#else
    for (int c = threadIdx.x; c < BM * 4; c += 256) {
      int row = c >> 2, col8 = (c & 3) * 8;
      *(v8bf*)&sA[buf][row * LDSW + col8] =
          *(const v8bf*)(A + (size_t)(blockM + row) * K + kk + col8);
    }
    for (int c = threadIdx.x; c < BN * 4; c += 256) {
      int row = c >> 2, col8 = (c & 3) * 8;
      *(v8bf*)&sB[buf][row * LDSW + col8] =
          *(const v8bf*)(BT + (size_t)(blockN + row) * K + kk + col8);
    }
#endif
  };

  stage(0, 0);
  int buf = 0;
  for (int kk = 0; kk < K; kk += BK) {
    const bool hasnext = (kk + BK) < K;
    if (hasnext) stage(buf ^ 1, kk + BK);   // DMA next slab while we compute this one
#ifdef USE_TDM
    if (wave == 0) {
      if (hasnext) __builtin_amdgcn_s_wait_tensorcnt(2);  // retire front A/B pair only
      else         __builtin_amdgcn_s_wait_tensorcnt(0);
    }
#endif
    __syncthreads();

    v16bf afrag[2], bfrag[4];
#pragma unroll
    for (int tm = 0; tm < 2; ++tm) {
      const bf16_t* p = &sA[buf][(wm * 32 + tm * 16 + l16) * LDSW + half * 8];
      v8bf lo = *(const v8bf*)p;            // ds_load_b128
      v8bf hi = *(const v8bf*)(p + 16);
#pragma unroll
      for (int i = 0; i < 8; ++i) { afrag[tm][i] = lo[i]; afrag[tm][i + 8] = hi[i]; }
    }
#pragma unroll
    for (int tn = 0; tn < 4; ++tn) {
      const bf16_t* p = &sB[buf][(wn * 64 + tn * 16 + l16) * LDSW + half * 8];
      v8bf lo = *(const v8bf*)p;
      v8bf hi = *(const v8bf*)(p + 16);
#pragma unroll
      for (int i = 0; i < 8; ++i) { bfrag[tn][i] = lo[i]; bfrag[tn][i + 8] = hi[i]; }
    }
#pragma unroll
    for (int tm = 0; tm < 2; ++tm)
#pragma unroll
      for (int tn = 0; tn < 4; ++tn)
        acc[tm][tn] = __builtin_amdgcn_wmma_f32_16x16x32_bf16(
            false, afrag[tm], false, bfrag[tn], (short)0, acc[tm][tn], false, false);

    __syncthreads();   // everyone done with buf before it is restaged
    buf ^= 1;
  }

#pragma unroll
  for (int tm = 0; tm < 2; ++tm)
#pragma unroll
    for (int tn = 0; tn < 4; ++tn) {
      int col = blockN + wn * 64 + tn * 16 + l16;
      float bv = bias ? bias[col] : 0.f;
#pragma unroll
      for (int r = 0; r < 8; ++r) {
        int row = blockM + wm * 32 + tm * 16 + r + half * 8;
        C[(size_t)row * N + col] = acc[tm][tn][r] + bv;
      }
    }
}

// ---------------------------------------------------------------------------
// RoPE + scale + pack: qkv f32 [S][2560] ->
//   qb [16][S][128] bf16 (pre-scaled by 1/sqrt(128)), kb [2][S][128] bf16,
//   vT [2][128][S] bf16 (transposed so PV B-fragments are contiguous)
__global__ __launch_bounds__(256) void k_rope_pack(const float* __restrict__ qkv,
                                                   const float* __restrict__ cosb,
                                                   const float* __restrict__ sinb,
                                                   bf16_t* __restrict__ qb,
                                                   bf16_t* __restrict__ kb,
                                                   bf16_t* __restrict__ vT) {
  const int s = blockIdx.x;
  const float* row = qkv + (size_t)s * QKV_N;
  const float scale = 0.08838834764831845f;  // 128^-0.5
  const float* cs = cosb + (size_t)s * 64;
  const float* sn = sinb + (size_t)s * 64;

  for (int i = threadIdx.x; i < H_Q * 64; i += 256) {
    int hh = i >> 6, d = i & 63;
    float x1 = row[hh * HD_ + d], x2 = row[hh * HD_ + 64 + d];
    float c = cs[d], si = sn[d];
    size_t base = ((size_t)hh * S_LEN + s) * HD_;
    qb[base + d]      = (bf16_t)((x1 * c - x2 * si) * scale);
    qb[base + 64 + d] = (bf16_t)((x1 * si + x2 * c) * scale);
  }
  for (int i = threadIdx.x; i < H_KV * 64; i += 256) {
    int kvh = i >> 6, d = i & 63;
    const float* kr = row + H_Q * HD_ + kvh * HD_;
    float x1 = kr[d], x2 = kr[64 + d];
    float c = cs[d], si = sn[d];
    size_t base = ((size_t)kvh * S_LEN + s) * HD_;
    kb[base + d]      = (bf16_t)(x1 * c - x2 * si);
    kb[base + 64 + d] = (bf16_t)(x1 * si + x2 * c);
  }
  for (int i = threadIdx.x; i < H_KV * HD_; i += 256) {
    int kvh = i >> 7, d = i & 127;
    float v = row[(H_Q + H_KV) * HD_ + kvh * HD_ + d];
    vT[((size_t)kvh * HD_ + d) * S_LEN + s] = (bf16_t)v;
  }
}

// ---------------------------------------------------------------------------
// Causal flash attention. One wave owns one 16-row Q tile, head = blockIdx.y.
__global__ __launch_bounds__(256) void k_attention(const bf16_t* __restrict__ qb,
                                                   const bf16_t* __restrict__ kb,
                                                   const bf16_t* __restrict__ vT,
                                                   bf16_t* __restrict__ attnb) {
  __shared__ __align__(16) bf16_t pstage[8][16][32];

  const int lane = threadIdx.x & 31;
  const int wave = threadIdx.x >> 5;
  const int half = lane >> 4, l16 = lane & 15;
  const int h  = blockIdx.y;
  const int kv = h >> 3;                       // nrep = 8
  const int q0 = (blockIdx.x * 8 + wave) * 16;

  const bf16_t* Q  = qb + ((size_t)h * S_LEN + q0) * HD_;
  const bf16_t* Kb = kb + (size_t)kv * S_LEN * HD_;
  const bf16_t* Vt = vT + (size_t)kv * HD_ * S_LEN;

  v16bf qfrag[4];
#pragma unroll
  for (int kk = 0; kk < 4; ++kk) {
    const bf16_t* p = Q + (size_t)l16 * HD_ + kk * 32 + half * 8;
    v8bf lo = *(const v8bf*)p;
    v8bf hi = *(const v8bf*)(p + 16);
#pragma unroll
    for (int i = 0; i < 8; ++i) { qfrag[kk][i] = lo[i]; qfrag[kk][i + 8] = hi[i]; }
  }

  float m[8], l[8];
  v8f o[8];
#pragma unroll
  for (int r = 0; r < 8; ++r) { m[r] = NEGF; l[r] = 0.f; }
#pragma unroll
  for (int n = 0; n < 8; ++n) o[n] = (v8f){0.f,0.f,0.f,0.f,0.f,0.f,0.f,0.f};

  const int nsteps = (q0 + 47) >> 5;  // keys [0, q0+16), step 32
  for (int j = 0; j < nsteps; ++j) {
    const int j0 = j << 5;

    v8f sc[2];
#pragma unroll
    for (int t = 0; t < 2; ++t) {
      sc[t] = (v8f){0.f,0.f,0.f,0.f,0.f,0.f,0.f,0.f};
#pragma unroll
      for (int kk = 0; kk < 4; ++kk) {
        const bf16_t* p = Kb + (size_t)(j0 + t * 16 + l16) * HD_ + kk * 32 + half * 8;
        v8bf lo = *(const v8bf*)p;
        v8bf hi = *(const v8bf*)(p + 16);
        v16bf kf;
#pragma unroll
        for (int i = 0; i < 8; ++i) { kf[i] = lo[i]; kf[i + 8] = hi[i]; }
        sc[t] = __builtin_amdgcn_wmma_f32_16x16x32_bf16(
            false, qfrag[kk], false, kf, (short)0, sc[t], false, false);
      }
    }

    float pv[2][8];
#pragma unroll
    for (int r = 0; r < 8; ++r) {
      int qrow = q0 + r + half * 8;
      float tmp[2], mrow = NEGF;
#pragma unroll
      for (int t = 0; t < 2; ++t) {
        int kcol = j0 + t * 16 + l16;
        tmp[t] = (kcol > qrow) ? NEGF : sc[t][r];
        mrow = fmaxf(mrow, tmp[t]);
      }
#pragma unroll
      for (int msk = 8; msk >= 1; msk >>= 1)
        mrow = fmaxf(mrow, __shfl_xor(mrow, msk, 32));
      float mnew = fmaxf(m[r], mrow);
      float csc  = __expf(m[r] - mnew);
      float rs = 0.f;
#pragma unroll
      for (int t = 0; t < 2; ++t) {
        float p = (tmp[t] <= NEGF) ? 0.f : __expf(tmp[t] - mnew);
        pv[t][r] = p;
        rs += p;
      }
#pragma unroll
      for (int msk = 8; msk >= 1; msk >>= 1)
        rs += __shfl_xor(rs, msk, 32);
      l[r] = l[r] * csc + rs;
      m[r] = mnew;
#pragma unroll
      for (int n = 0; n < 8; ++n) o[n][r] *= csc;
    }

    // restage P (C layout -> A layout) through wave-private LDS
#pragma unroll
    for (int t = 0; t < 2; ++t)
#pragma unroll
      for (int r = 0; r < 8; ++r)
        pstage[wave][r + half * 8][t * 16 + l16] = (bf16_t)pv[t][r];
    asm volatile("" ::: "memory");  // keep DS stores before DS loads (HW in-order per wave)
    v16bf pfrag;
    {
      const bf16_t* pp = &pstage[wave][l16][half * 8];
      v8bf lo = *(const v8bf*)pp;
      v8bf hi = *(const v8bf*)(pp + 16);
#pragma unroll
      for (int i = 0; i < 8; ++i) { pfrag[i] = lo[i]; pfrag[i + 8] = hi[i]; }
    }

#pragma unroll
    for (int n = 0; n < 8; ++n) {
      const bf16_t* p = Vt + (size_t)(n * 16 + l16) * S_LEN + j0 + half * 8;
      v8bf lo = *(const v8bf*)p;
      v8bf hi = *(const v8bf*)(p + 16);
      v16bf vf;
#pragma unroll
      for (int i = 0; i < 8; ++i) { vf[i] = lo[i]; vf[i + 8] = hi[i]; }
      o[n] = __builtin_amdgcn_wmma_f32_16x16x32_bf16(
          false, pfrag, false, vf, (short)0, o[n], false, false);
    }
  }

  float inv[8];
#pragma unroll
  for (int r = 0; r < 8; ++r) inv[r] = 1.f / l[r];
#pragma unroll
  for (int n = 0; n < 8; ++n)
#pragma unroll
    for (int r = 0; r < 8; ++r) {
      int row = q0 + r + half * 8;
      attnb[(size_t)row * D_DIM + h * HD_ + n * 16 + l16] = (bf16_t)(o[n][r] * inv[r]);
    }
}

// ---------------------------------------------------------------------------
extern "C" void kernel_launch(void* const* d_in, const int* in_sizes, int n_in,
                              void* d_out, int out_size, void* d_ws, size_t ws_size,
                              hipStream_t stream) {
  const float* hs   = (const float*)d_in[0];
  const float* fcos = (const float*)d_in[1];
  const float* fsin = (const float*)d_in[2];
  const float* Wqkv = (const float*)d_in[7];
  const float* bqkv = (const float*)d_in[8];
  const float* Wo   = (const float*)d_in[9];
  float* out = (float*)d_out;

  char* ws = (char*)d_ws;
  size_t o_hsb   = 0;
  size_t o_wqkvT = o_hsb   + (size_t)S_LEN * D_DIM * 2;
  size_t o_woT   = o_wqkvT + (size_t)QKV_N * D_DIM * 2;
  size_t o_qkvf  = o_woT   + (size_t)D_DIM * D_DIM * 2;
  size_t o_qb    = o_qkvf  + (size_t)S_LEN * QKV_N * 4;
  size_t o_kb    = o_qb    + (size_t)H_Q  * S_LEN * HD_ * 2;
  size_t o_vT    = o_kb    + (size_t)H_KV * S_LEN * HD_ * 2;
  size_t o_attnb = o_vT    + (size_t)H_KV * HD_ * S_LEN * 2;

  bf16_t* hsb   = (bf16_t*)(ws + o_hsb);
  bf16_t* wqkvT = (bf16_t*)(ws + o_wqkvT);
  bf16_t* woT   = (bf16_t*)(ws + o_woT);
  float*  qkvf  = (float*) (ws + o_qkvf);
  bf16_t* qb    = (bf16_t*)(ws + o_qb);
  bf16_t* kbuf  = (bf16_t*)(ws + o_kb);
  bf16_t* vT    = (bf16_t*)(ws + o_vT);
  bf16_t* attnb = (bf16_t*)(ws + o_attnb);

  k_f32_to_bf16<<<1024, 256, 0, stream>>>(hs, hsb, S_LEN * D_DIM);
  k_transpose_bf16<<<1024, 256, 0, stream>>>(Wqkv, wqkvT, D_DIM, QKV_N);
  k_transpose_bf16<<<1024, 256, 0, stream>>>(Wo, woT, D_DIM, D_DIM);

  k_gemm_bf16<<<dim3(QKV_N / BN, S_LEN / BM), 256, 0, stream>>>(
      hsb, wqkvT, bqkv, qkvf, S_LEN, QKV_N, D_DIM);

  k_rope_pack<<<S_LEN, 256, 0, stream>>>(qkvf, fcos, fsin, qb, kbuf, vT);

  k_attention<<<dim3(S_LEN / (16 * 8), H_Q), 256, 0, stream>>>(qb, kbuf, vT, attnb);

  k_gemm_bf16<<<dim3(D_DIM / BN, S_LEN / BM), 256, 0, stream>>>(
      attnb, woT, nullptr, out, S_LEN, D_DIM, D_DIM);
}
